// EISEN_71485435675204
// MI455X (gfx1250) — matrix operand order repaired and checked by reading, not compiled, not run on hardware
//
#include <hip/hip_runtime.h>
#include <hip/hip_bf16.h>

// Problem constants (from the reference)
#define B_    2
#define N_    16384      // Hc*Wc
#define CIN_  256
#define CK_   32
#define S_    1024

typedef __attribute__((ext_vector_type(16))) _Float16 v16h;
typedef __attribute__((ext_vector_type(8)))  float    v8f;

// ---------------------------------------------------------------------------
// Kernel 1: feat = features . W_feat^T + b_feat  (M=B*N, K=256, N=32)
//           keyv = feat . W_key^T + b_key ; query = feat . W_query^T + b_query
// One wave (32 lanes) per 16-row tile; WMMA 16x16x32 f16->f32.
// ---------------------------------------------------------------------------
__global__ __launch_bounds__(256) void eisen_qk_wmma(
    const float* __restrict__ features,
    const float* __restrict__ W_feat,  const float* __restrict__ b_feat,
    const float* __restrict__ W_key,   const float* __restrict__ b_key,
    const float* __restrict__ W_query, const float* __restrict__ b_query,
    float* __restrict__ keyv_ws, float* __restrict__ query_ws)
{
    __shared__ _Float16 sfeat[8][16][CK_];   // per-wave 16x32 feat tile (1KB each)

    const int lane = threadIdx.x & 31;
    const int wv   = threadIdx.x >> 5;
    const int hl   = lane >> 4;     // half-wave select
    const int r    = lane & 15;     // row (A) / column (B,C,D) index
    const int tile = blockIdx.x * 8 + wv;
    const int m0   = tile * 16;                 // first row of this tile in B*N
    const int b    = m0 / N_;
    const int n0   = m0 - b * N_;               // N_ % 16 == 0 -> tile within one batch
    const float* fbase = features + (size_t)b * CIN_ * N_ + n0;

    // ---- Phase A: feat = features^T x W_feat^T over K=256 (8 steps x 2 col blocks)
    v8f acc0 = {}; v8f acc1 = {};
    #pragma unroll
    for (int k0 = 0; k0 < CIN_; k0 += 32) {
        // A fragment: 16x32 f16; lane<16 row=r holds K 0..15, lane>=16 K 16..31
        v16h a;
        #pragma unroll
        for (int e = 0; e < 16; ++e) {
            int c = k0 + hl * 16 + e;
            a[e] = (_Float16)fbase[(size_t)c * N_ + r];   // features[b][c][n0+r]
        }
        // B fragments: B[k][n] = W_feat[n][k]; lanes 0-15 hold K=0..15, 16-31 hold K=16..31
        v16h b0, b1;
        #pragma unroll
        for (int e = 0; e < 16; ++e) {
            int c = k0 + hl * 16 + e;
            b0[e] = (_Float16)W_feat[r * CIN_ + c];
            b1[e] = (_Float16)W_feat[(16 + r) * CIN_ + c];
        }
        acc0 = __builtin_amdgcn_wmma_f32_16x16x32_f16(false, a, false, b0, (short)0, acc0, false, false);
        acc1 = __builtin_amdgcn_wmma_f32_16x16x32_f16(false, a, false, b1, (short)0, acc1, false, false);
    }

    // C fragment -> LDS (row-major 16x32 f16) with bias; C: VGPR j = row (j + hl*8), col = r
    #pragma unroll
    for (int j = 0; j < 8; ++j) {
        int row = j + hl * 8;
        sfeat[wv][row][r]      = (_Float16)(acc0[j] + b_feat[r]);
        sfeat[wv][row][16 + r] = (_Float16)(acc1[j] + b_feat[16 + r]);
    }
    __syncthreads();

    // ---- Phase B: keyv / query (K=32, one WMMA step, two 16-col blocks each)
    v16h a2;
    #pragma unroll
    for (int e = 0; e < 16; ++e) {
        int c = hl * 16 + e;
        a2[e] = sfeat[wv][r][c];
    }
    v16h bk0, bk1, bq0, bq1;
    #pragma unroll
    for (int e = 0; e < 16; ++e) {
        int c = hl * 16 + e;
        bk0[e] = (_Float16)W_key  [r * CK_ + c];
        bk1[e] = (_Float16)W_key  [(16 + r) * CK_ + c];
        bq0[e] = (_Float16)W_query[r * CK_ + c];
        bq1[e] = (_Float16)W_query[(16 + r) * CK_ + c];
    }
    v8f z = {};
    v8f kc0 = __builtin_amdgcn_wmma_f32_16x16x32_f16(false, a2, false, bk0, (short)0, z, false, false);
    v8f kc1 = __builtin_amdgcn_wmma_f32_16x16x32_f16(false, a2, false, bk1, (short)0, z, false, false);
    v8f qc0 = __builtin_amdgcn_wmma_f32_16x16x32_f16(false, a2, false, bq0, (short)0, z, false, false);
    v8f qc1 = __builtin_amdgcn_wmma_f32_16x16x32_f16(false, a2, false, bq1, (short)0, z, false, false);

    #pragma unroll
    for (int j = 0; j < 8; ++j) {
        size_t mrow = (size_t)(m0 + j + hl * 8);
        keyv_ws [mrow * CK_ + r]       = kc0[j] + b_key[r];
        keyv_ws [mrow * CK_ + 16 + r]  = kc1[j] + b_key[16 + r];
        query_ws[mrow * CK_ + r]       = qc0[j] + b_query[r];
        query_ws[mrow * CK_ + 16 + r]  = qc1[j] + b_query[16 + r];
    }
}

// ---------------------------------------------------------------------------
// Wave reductions (wave32)
// ---------------------------------------------------------------------------
__device__ __forceinline__ float wave_sum(float v) {
    #pragma unroll
    for (int m = 16; m >= 1; m >>= 1) v += __shfl_xor(v, m, 32);
    return v;
}
__device__ __forceinline__ float wave_max(float v) {
    #pragma unroll
    for (int m = 16; m >= 1; m >>= 1) v = fmaxf(v, __shfl_xor(v, m, 32));
    return v;
}

// ---------------------------------------------------------------------------
// Kernel 2: one wave per (b,n) row. Gather-dot logits (keyv rows are L2
// resident: 4MB << 192MB L2), fused masked softmax + KL loss.
// ---------------------------------------------------------------------------
__global__ __launch_bounds__(256) void eisen_affinity_loss(
    const float* __restrict__ query_ws, const float* __restrict__ keyv_ws,
    const int*  __restrict__ sample_inds, const int* __restrict__ segment,
    float* __restrict__ logits, float* __restrict__ acc)
{
    const int lane = threadIdx.x & 31;
    const int wv   = threadIdx.x >> 5;
    const int row  = blockIdx.x * 8 + wv;       // b*N + n
    const int b    = row / N_;

    // Query row is wave-uniform: keep in registers (hardware coalesces the broadcast)
    const float4* qp = (const float4*)(query_ws + (size_t)row * CK_);
    float4 q[8];
    #pragma unroll
    for (int w = 0; w < 8; ++w) q[w] = qp[w];

    const int   seg   = segment[row];
    const float mask  = (seg != 0) ? 1.0f : 0.0f;
    const int*  idxrow = sample_inds + (size_t)row * S_;
    const int*  segb   = segment + b * N_;
    const float* keyb  = keyv_ws + (size_t)b * N_ * CK_;
    float* outrow = logits + (size_t)row * S_;
    const float scale = 0.17677669529663687f;   // CK^-0.5

    float lg[32];                // full 1024-logit row lives in registers (32/lane)
    float lmax = -3.0e38f;
    unsigned tgtbits = 0;
    float tcnt = 0.0f;

    #pragma unroll
    for (int it = 0; it < 32; ++it) {
        int s  = it * 32 + lane;
        int id = idxrow[s];
        __builtin_prefetch(idxrow + ((it + 4) & 31) * 32 + lane, 0, 0);
        const float4* kr = (const float4*)(keyb + (size_t)id * CK_);
        float d = 0.0f;
        #pragma unroll
        for (int w = 0; w < 8; ++w) {
            float4 kv = kr[w];
            d += q[w].x * kv.x + q[w].y * kv.y + q[w].z * kv.z + q[w].w * kv.w;
        }
        float lv = d * scale;
        lg[it] = lv;
        outrow[s] = lv;
        lmax = fmaxf(lmax, lv);
        if (segb[id] == seg) { tgtbits |= (1u << it); tcnt += 1.0f; }
    }

    lmax = wave_max(lmax);
    float sumtgt = wave_sum(tcnt);
    float sume = 0.0f;
    #pragma unroll
    for (int it = 0; it < 32; ++it) sume += __expf(lg[it] - lmax);
    sume = wave_sum(sume);

    float total_e = mask * sume;                 // e[s] = exp(l-m)*mask uniformly
    float tt    = 1.0f / (sumtgt + 1e-9f);       // y_true value at tgt==1 entries
    float tlogt = tt * __logf(tt);               // xlogy(t,t)

    float kl = 0.0f;
    #pragma unroll
    for (int it = 0; it < 32; ++it) {
        float e  = __expf(lg[it] - lmax) * mask;
        float p  = e / (total_e + 1e-9f);
        float yp = __logf(fmaxf(p, 1e-8f));
        if ((tgtbits >> it) & 1u) kl += tlogt - tt * yp;
    }
    kl = wave_sum(kl) * mask;

    if (lane == 0) {
        atomicAdd(&acc[0], kl);     // loss numerator
        atomicAdd(&acc[1], mask);   // agg (valid-row count)
    }
}

// ---------------------------------------------------------------------------
// Kernel 3: finalize scalar loss
// ---------------------------------------------------------------------------
__global__ void eisen_finalize(const float* __restrict__ acc, float* __restrict__ loss_out)
{
    loss_out[0] = acc[0] / (acc[1] + 1e-9f);
}

// ---------------------------------------------------------------------------
extern "C" void kernel_launch(void* const* d_in, const int* in_sizes, int n_in,
                              void* d_out, int out_size, void* d_ws, size_t ws_size,
                              hipStream_t stream)
{
    (void)in_sizes; (void)n_in; (void)out_size; (void)ws_size;

    const float* features    = (const float*)d_in[0];
    const int*   segment     = (const int*)  d_in[1];
    const int*   sample_inds = (const int*)  d_in[2];
    const float* W_feat      = (const float*)d_in[3];
    const float* b_feat      = (const float*)d_in[4];
    const float* W_key       = (const float*)d_in[5];
    const float* b_key       = (const float*)d_in[6];
    const float* W_query     = (const float*)d_in[7];
    const float* b_query     = (const float*)d_in[8];

    float* logits   = (float*)d_out;                          // (B,N,S)
    float* loss_out = logits + (size_t)B_ * N_ * S_;          // scalar after logits

    float* keyv_ws  = (float*)d_ws;                           // 4 MB
    float* query_ws = keyv_ws + (size_t)B_ * N_ * CK_;        // 4 MB
    float* acc      = query_ws + (size_t)B_ * N_ * CK_;       // 2 floats

    hipMemsetAsync(acc, 0, 2 * sizeof(float), stream);

    // Kernel 1: 2048 16-row tiles, 8 waves (tiles) per 256-thread block
    eisen_qk_wmma<<<(B_ * N_ / 16) / 8, 256, 0, stream>>>(
        features, W_feat, b_feat, W_key, b_key, W_query, b_query,
        keyv_ws, query_ws);

    // Kernel 2: 32768 rows, 8 waves (rows) per block
    eisen_affinity_loss<<<(B_ * N_) / 8, 256, 0, stream>>>(
        query_ws, keyv_ws, sample_inds, segment, logits, acc);

    eisen_finalize<<<1, 1, 0, stream>>>(acc, loss_out);
}
